// PolicyValueNet_20555713479229
// MI455X (gfx1250) — matrix operand order repaired and checked by reading, not compiled, not run on hardware
//
#include <hip/hip_runtime.h>
#include <hip/hip_bf16.h>

typedef __attribute__((ext_vector_type(16))) __bf16 v16bf;
typedef __attribute__((ext_vector_type(8)))  float  v8f;

// Problem constants (match reference)
static constexpr int Bc  = 2048;
static constexpr int NNc = 64;
static constexpr int Dc  = 128;
static constexpr int HDc = 128;
static constexpr int Lc  = 4;
static constexpr int Hc  = 8;
static constexpr int Sc  = 8;
static constexpr int FINc = 16;
static constexpr int Ac  = 56;                  // S*(S-1)
static constexpr long Nn = (long)Bc * NNc;      // 131072 nodes
static constexpr long Ec = (long)Bc * 8 * NNc;  // 1,048,576 edges

#define DEV __device__ __forceinline__

// ---------------------------------------------------------------------------
// A fragment loader (bf16 16x32 MxK from f32 row-major), fully compile-time K.
// lane&15 = row. lanes<16 hold K=k0+{0..7,16..23}; lanes>=16 K=k0+{8..15,24..31}.
// Both 8-float runs are contiguous; run1 is valid iff K%32==0 (true for all
// our K in {16,128,256} except the K=16 pad case, decided at compile time).
template <int K>
DEV v16bf load_a_frag(const float* __restrict__ A, int lda, int m0, int k0,
                      float scale, int lane) {
  int row  = m0 + (lane & 15);
  int koff = k0 + ((lane & 16) ? 8 : 0);
  const float* p = A + (long)row * lda + koff;
  v16bf a;
  float4 q0 = *(const float4*)(p);
  float4 q1 = *(const float4*)(p + 4);
  a[0] = (__bf16)(q0.x * scale); a[1] = (__bf16)(q0.y * scale);
  a[2] = (__bf16)(q0.z * scale); a[3] = (__bf16)(q0.w * scale);
  a[4] = (__bf16)(q1.x * scale); a[5] = (__bf16)(q1.y * scale);
  a[6] = (__bf16)(q1.z * scale); a[7] = (__bf16)(q1.w * scale);
  if constexpr (K % 32 == 0) {
    float4 q2 = *(const float4*)(p + 16);
    float4 q3 = *(const float4*)(p + 20);
    a[8]  = (__bf16)(q2.x * scale); a[9]  = (__bf16)(q2.y * scale);
    a[10] = (__bf16)(q2.z * scale); a[11] = (__bf16)(q2.w * scale);
    a[12] = (__bf16)(q3.x * scale); a[13] = (__bf16)(q3.y * scale);
    a[14] = (__bf16)(q3.z * scale); a[15] = (__bf16)(q3.w * scale);
  } else {
#pragma unroll
    for (int e = 8; e < 16; ++e) a[e] = (__bf16)0.0f;
  }
  return a;
}

// ---------------------------------------------------------------------------
// Weight pack: f32 [K x 128] row-major -> bf16 fragment-major
// layout: [ktile][ntile(8)][lane(32)][e(16)], tile = 512 bf16, ktile = 4096.
// gridDim.y = head (or layer) index.
__global__ void pack_b_kernel(const float* __restrict__ W, long wHeadOff, int K,
                              __bf16* __restrict__ out, long outHeadOff) {
  int head = blockIdx.y;
  const float* Wh = W + (long)head * wHeadOff;
  __bf16* oh = out + (long)head * outHeadOff;
  int ktiles = (K + 31) / 32;
  long total = (long)ktiles * 8 * 512;
  long stride = (long)gridDim.x * blockDim.x;
  for (long gid = (long)blockIdx.x * blockDim.x + threadIdx.x; gid < total;
       gid += stride) {
    int e    = (int)(gid & 15);
    int lane = (int)((gid >> 4) & 31);
    int nt   = (int)((gid >> 9) & 7);
    int kt   = (int)(gid >> 12);
    int k    = kt * 32 + ((lane & 16) ? 8 : 0) + (e & 7) + ((e & 8) ? 16 : 0);
    int col  = nt * 16 + (lane & 15);
    oh[gid] = (__bf16)((k < K) ? Wh[(long)k * 128 + col] : 0.0f);
  }
}

// ---------------------------------------------------------------------------
// Generic WMMA GEMM: C[M x 128] = act(A[M x K] @ W[K x 128] + bias)
// grid.x = M/16 row tiles, grid.y = heads; 256 threads = 8 waves = 8 col tiles
template <int K, int ACT>
__global__ __launch_bounds__(256) void wmma_gemm_packed(
    const float* __restrict__ A, long aHeadOff, int lda,
    const __bf16* __restrict__ Bp, long bHeadOff,
    const float* __restrict__ bias, long biasHeadOff,
    float* __restrict__ C, long cHeadOff, int ldc) {
  constexpr int KT = (K + 31) / 32;
  int head = blockIdx.y;
  const float*  Ah = A + (long)head * aHeadOff;
  const __bf16* Bh = Bp + (long)head * bHeadOff;
  const float*  bh = bias + (long)head * biasHeadOff;
  float*        Ch = C + (long)head * cHeadOff;

  int lane = threadIdx.x & 31;
  int nt   = threadIdx.x >> 5;
  int m0   = blockIdx.x * 16;
  const __bf16* bfrag = Bh + (long)nt * 512 + lane * 16;

  v8f c = {};
#pragma unroll
  for (int kt = 0; kt < KT; ++kt) {
    v16bf a = load_a_frag<K>(Ah, lda, m0, kt * 32, 1.0f, lane);
    v16bf b = *(const v16bf*)(bfrag + (long)kt * 4096);
    c = __builtin_amdgcn_wmma_f32_16x16x32_bf16(false, a, false, b,
                                                (short)0, c, false, false);
  }

  int col   = nt * 16 + (lane & 15);
  int rbase = m0 + ((lane & 16) ? 8 : 0);
  float bv  = bh[col];
#pragma unroll
  for (int v = 0; v < 8; ++v) {
    float val = c[v] + bv;
    if (ACT) val = fmaxf(val, 0.0f);
    Ch[(long)(rbase + v) * ldc + col] = val;
  }
}

// ---------------------------------------------------------------------------
// GNN layer: hout = relu(h @ Wself + (agg * invdeg[row]) @ Wnbr + bias)
__global__ __launch_bounds__(256) void wmma_layer_packed(
    const float* __restrict__ h, const float* __restrict__ agg,
    const float* __restrict__ invdeg,
    const __bf16* __restrict__ pSelf, const __bf16* __restrict__ pNbr,
    const float* __restrict__ bias, float* __restrict__ hout) {
  int lane = threadIdx.x & 31;
  int nt   = threadIdx.x >> 5;
  int m0   = blockIdx.x * 16;
  float sc = invdeg[m0 + (lane & 15)];
  const __bf16* bs = pSelf + (long)nt * 512 + lane * 16;
  const __bf16* bn = pNbr  + (long)nt * 512 + lane * 16;

  v8f c = {};
#pragma unroll
  for (int kt = 0; kt < 4; ++kt) {
    v16bf a = load_a_frag<128>(h, Dc, m0, kt * 32, 1.0f, lane);
    v16bf b = *(const v16bf*)(bs + (long)kt * 4096);
    c = __builtin_amdgcn_wmma_f32_16x16x32_bf16(false, a, false, b,
                                                (short)0, c, false, false);
  }
#pragma unroll
  for (int kt = 0; kt < 4; ++kt) {
    v16bf a = load_a_frag<128>(agg, Dc, m0, kt * 32, sc, lane);
    v16bf b = *(const v16bf*)(bn + (long)kt * 4096);
    c = __builtin_amdgcn_wmma_f32_16x16x32_bf16(false, a, false, b,
                                                (short)0, c, false, false);
  }

  int col   = nt * 16 + (lane & 15);
  int rbase = m0 + ((lane & 16) ? 8 : 0);
  float bv  = bias[col];
#pragma unroll
  for (int v = 0; v < 8; ++v)
    hout[(long)(rbase + v) * Dc + col] = fmaxf(c[v] + bv, 0.0f);
}

// ---------------------------------------------------------------------------
// Scalar helper kernels
__global__ void zero_kernel(float* __restrict__ p, long n) {
  long i = (long)blockIdx.x * blockDim.x + threadIdx.x;
  long stride = (long)gridDim.x * blockDim.x;
  for (; i < n; i += stride) p[i] = 0.0f;
}

__global__ void deg_kernel(const int* __restrict__ dst, float* __restrict__ deg) {
  long e = (long)blockIdx.x * blockDim.x + threadIdx.x;
  if (e >= Ec) return;
  atomicAdd(&deg[dst[e]], 1.0f);
}

__global__ void recip_kernel(float* __restrict__ d) {
  long i = (long)blockIdx.x * blockDim.x + threadIdx.x;
  if (i >= Nn) return;
  d[i] = 1.0f / fmaxf(d[i], 1.0f);
}

// 32 threads per edge, 4 floats each: agg[dst] += h[src]
__global__ void scatter_kernel(const float* __restrict__ h,
                               const int* __restrict__ src,
                               const int* __restrict__ dst,
                               float* __restrict__ agg) {
  long gid = (long)blockIdx.x * blockDim.x + threadIdx.x;
  long e   = gid >> 5;
  int part = (int)(gid & 31);
  if (e >= Ec) return;
  int s = src[e], d = dst[e];
  const float* hs = h + (long)s * Dc + part * 4;
  float*       ad = agg + (long)d * Dc + part * 4;
  float4 q = *(const float4*)hs;
  atomicAdd(&ad[0], q.x);
  atomicAdd(&ad[1], q.y);
  atomicAdd(&ad[2], q.z);
  atomicAdd(&ad[3], q.w);
}

// pooled[b,h,0:D]=mean over subset nodes; pooled[b,h,D:2D]=max
__global__ void pool_kernel(const float* __restrict__ he,
                            const int* __restrict__ subsets,
                            float* __restrict__ pooled) {
  long gid = (long)blockIdx.x * blockDim.x + threadIdx.x;
  if (gid >= (long)Bc * Hc * Dc) return;
  int d = (int)(gid % Dc);
  int h = (int)((gid / Dc) % Hc);
  long b = gid / ((long)Dc * Hc);
  float sum = 0.0f, mx = -INFINITY;
#pragma unroll
  for (int s = 0; s < Sc; ++s) {
    int node = subsets[h * Sc + s];
    float v = he[(b * NNc + node) * Dc + d];
    sum += v;
    mx = fmaxf(mx, v);
  }
  float* row = pooled + (b * Hc + h) * (2 * Dc);
  row[d]      = sum * (1.0f / Sc);
  row[Dc + d] = mx;
}

// LayerNorm + relu per (b,h) row of 128
__global__ void ln_kernel(const float* __restrict__ z,
                          const float* __restrict__ g,
                          const float* __restrict__ bt,
                          float* __restrict__ out) {
  long r = (long)blockIdx.x * blockDim.x + threadIdx.x;
  if (r >= (long)Bc * Hc) return;
  int h = (int)(r % Hc);
  const float* zp = z + r * HDc;
  float mu = 0.0f;
#pragma unroll 8
  for (int d = 0; d < HDc; ++d) mu += zp[d];
  mu *= (1.0f / HDc);
  float var = 0.0f;
#pragma unroll 8
  for (int d = 0; d < HDc; ++d) { float t = zp[d] - mu; var += t * t; }
  var *= (1.0f / HDc);
  float inv = rsqrtf(var + 1e-5f);
  const float* gp = g + h * HDc;
  const float* bp = bt + h * HDc;
  float* op = out + r * HDc;
#pragma unroll 8
  for (int d = 0; d < HDc; ++d)
    op[d] = fmaxf((zp[d] - mu) * inv * gp[d] + bp[d], 0.0f);
}

// ctxin[b,d] = mean over heads of hemb[b,h,d]
__global__ void ctxmean_kernel(const float* __restrict__ hemb,
                               float* __restrict__ ctxin) {
  long gid = (long)blockIdx.x * blockDim.x + threadIdx.x;
  if (gid >= (long)Bc * HDc) return;
  int d = (int)(gid % HDc);
  long b = gid / HDc;
  float s = 0.0f;
#pragma unroll
  for (int h = 0; h < Hc; ++h) s += hemb[(b * Hc + h) * HDc + d];
  ctxin[gid] = s * (1.0f / Hc);
}

// pc[b,h,f] = f<HD ? hemb[b,h,f] : ctx[b,f-HD]
__global__ void pc_kernel(const float* __restrict__ hemb,
                          const float* __restrict__ ctx,
                          float* __restrict__ pc) {
  long gid = (long)blockIdx.x * blockDim.x + threadIdx.x;
  if (gid >= (long)Bc * Hc * 2 * HDc) return;
  int f = (int)(gid % (2 * HDc));
  long bh = gid / (2 * HDc);
  long b = bh / Hc;
  pc[gid] = (f < HDc) ? hemb[bh * HDc + f] : ctx[b * HDc + (f - HDc)];
}

// logits[b,h,a] = hh[b,h,:] . pol_W2[h,:,a] + pol_b2[h,a]
__global__ void logits_kernel(const float* __restrict__ hh,
                              const float* __restrict__ W2,
                              const float* __restrict__ b2,
                              float* __restrict__ out) {
  long gid = (long)blockIdx.x * blockDim.x + threadIdx.x;
  if (gid >= (long)Bc * Hc * Ac) return;
  int a = (int)(gid % Ac);
  int h = (int)((gid / Ac) % Hc);
  long b = gid / ((long)Ac * Hc);
  const float* hp = hh + (b * Hc + h) * HDc;
  const float* wp = W2 + (long)h * HDc * Ac + a;
  float s = b2[h * Ac + a];
#pragma unroll 8
  for (int d = 0; d < HDc; ++d) s += hp[d] * wp[(long)d * Ac];
  out[gid] = s;
}

// gm[b,d] = mean over NN nodes of he[b,n,d]
__global__ void gm_kernel(const float* __restrict__ he, float* __restrict__ gm) {
  long gid = (long)blockIdx.x * blockDim.x + threadIdx.x;
  if (gid >= (long)Bc * Dc) return;
  int d = (int)(gid % Dc);
  long b = gid / Dc;
  float s = 0.0f;
#pragma unroll 8
  for (int n = 0; n < NNc; ++n) s += he[(b * NNc + n) * Dc + d];
  gm[gid] = s * (1.0f / NNc);
}

// v[b] = tanh(t1[b,:] . val_W2 + val_b2)
__global__ void value_kernel(const float* __restrict__ t1,
                             const float* __restrict__ W2,
                             const float* __restrict__ b2,
                             float* __restrict__ out) {
  long b = (long)blockIdx.x * blockDim.x + threadIdx.x;
  if (b >= Bc) return;
  const float* tp = t1 + b * Dc;
  float s = b2[0];
#pragma unroll 8
  for (int d = 0; d < Dc; ++d) s += tp[d] * W2[d];
  out[b] = tanhf(s);
}

// ---------------------------------------------------------------------------
extern "C" void kernel_launch(void* const* d_in, const int* in_sizes, int n_in,
                              void* d_out, int out_size, void* d_ws, size_t ws_size,
                              hipStream_t stream) {
  const float* x       = (const float*)d_in[0];
  const int*   ei      = (const int*)d_in[1];
  const int*   subsets = (const int*)d_in[3];
  const float* W_in    = (const float*)d_in[4];
  const float* b_in    = (const float*)d_in[5];
  const float* W_self  = (const float*)d_in[6];
  const float* W_nbr   = (const float*)d_in[7];
  const float* b_enc   = (const float*)d_in[8];
  const float* ext_W   = (const float*)d_in[9];
  const float* ext_b   = (const float*)d_in[10];
  const float* ln_g    = (const float*)d_in[11];
  const float* ln_b    = (const float*)d_in[12];
  const float* hub_W   = (const float*)d_in[13];
  const float* hub_b   = (const float*)d_in[14];
  const float* pol_W1  = (const float*)d_in[15];
  const float* pol_b1  = (const float*)d_in[16];
  const float* pol_W2  = (const float*)d_in[17];
  const float* pol_b2  = (const float*)d_in[18];
  const float* val_W1  = (const float*)d_in[19];
  const float* val_b1  = (const float*)d_in[20];
  const float* val_W2  = (const float*)d_in[21];
  const float* val_b2  = (const float*)d_in[22];

  const int* src = ei;
  const int* dst = ei + Ec;
  float* out = (float*)d_out;

  // Workspace layout: f32 buffers
  float* ws = (float*)d_ws;
  float* hA     = ws;                 // N*D
  float* hB     = hA + Nn * Dc;       // N*D
  float* agg    = hB + Nn * Dc;       // N*D (reused for head stage)
  float* invdeg = agg + Nn * Dc;      // N
  float* ctxin  = invdeg + Nn;        // B*HD
  float* ctx    = ctxin + (long)Bc * HDc;
  float* gm     = ctx + (long)Bc * HDc;
  float* t1     = gm + (long)Bc * Dc;
  float* fend   = t1 + (long)Bc * Dc;
  // head-stage buffers overlaid on agg (used only after GNN layers)
  float* pooled = agg;                                  // B*H*2D
  float* zbuf   = pooled + (long)Bc * Hc * 2 * Dc;      // B*H*HD
  float* hemb   = zbuf + (long)Bc * Hc * HDc;           // B*H*HD
  float* pc     = hemb + (long)Bc * Hc * HDc;           // B*H*2HD
  float* hh     = pc + (long)Bc * Hc * 2 * HDc;         // B*H*HD

  // packed bf16 weight fragments (64B aligned)
  uintptr_t up = ((uintptr_t)fend + 63) & ~(uintptr_t)63;
  __bf16* pWin  = (__bf16*)up;             // 1*8*512  = 4096
  __bf16* pSelf = pWin + 4096;             // 4 layers * 16384
  __bf16* pNbr  = pSelf + 4 * 16384;       // 4 layers * 16384
  __bf16* pExt  = pNbr + 4 * 16384;        // 8 heads * 32768
  __bf16* pHub  = pExt + 8L * 32768;       // 16384
  __bf16* pPol  = pHub + 16384;            // 8 heads * 32768
  __bf16* pVal  = pPol + 8L * 32768;       // 16384

  const int T = 256;

  // 0. pack all weights into WMMA fragment layout (one-time, tiny)
  pack_b_kernel<<<dim3(16, 1), T, 0, stream>>>(W_in, 0, FINc, pWin, 0);
  pack_b_kernel<<<dim3(64, Lc), T, 0, stream>>>(W_self, (long)Dc * Dc, Dc,
                                                pSelf, 16384);
  pack_b_kernel<<<dim3(64, Lc), T, 0, stream>>>(W_nbr, (long)Dc * Dc, Dc,
                                                pNbr, 16384);
  pack_b_kernel<<<dim3(128, Hc), T, 0, stream>>>(ext_W, (long)2 * Dc * HDc,
                                                 2 * Dc, pExt, 32768);
  pack_b_kernel<<<dim3(64, 1), T, 0, stream>>>(hub_W, 0, HDc, pHub, 0);
  pack_b_kernel<<<dim3(128, Hc), T, 0, stream>>>(pol_W1, (long)2 * HDc * HDc,
                                                 2 * HDc, pPol, 32768);
  pack_b_kernel<<<dim3(64, 1), T, 0, stream>>>(val_W1, 0, Dc, pVal, 0);

  // 1. degrees
  zero_kernel<<<1024, T, 0, stream>>>(invdeg, Nn);
  deg_kernel<<<(int)(Ec / T), T, 0, stream>>>(dst, invdeg);
  recip_kernel<<<(int)(Nn / T), T, 0, stream>>>(invdeg);

  // 2. input projection: hA = relu(x @ W_in + b_in), K=16 (zero-padded to 32)
  wmma_gemm_packed<FINc, 1><<<dim3((int)(Nn / 16), 1), T, 0, stream>>>(
      x, 0, FINc, pWin, 0, b_in, 0, hA, 0, Dc);

  // 3. GNN layers
  float* hcur = hA;
  float* hnxt = hB;
  for (int l = 0; l < Lc; ++l) {
    zero_kernel<<<4096, T, 0, stream>>>(agg, Nn * Dc);
    scatter_kernel<<<(int)(Ec * 32 / T), T, 0, stream>>>(hcur, src, dst, agg);
    wmma_layer_packed<<<(int)(Nn / 16), T, 0, stream>>>(
        hcur, agg, invdeg, pSelf + (long)l * 16384, pNbr + (long)l * 16384,
        b_enc + (long)l * Dc, hnxt);
    float* tmp = hcur; hcur = hnxt; hnxt = tmp;
  }
  const float* he = hcur;   // after 4 swaps this is hA

  // 4. pooling -> pooled[B,H,2D]
  pool_kernel<<<(int)((long)Bc * Hc * Dc / T), T, 0, stream>>>(he, subsets, pooled);

  // 5. z = pooled @ ext_W + ext_b   (per-head, K=256)
  wmma_gemm_packed<2 * Dc, 0><<<dim3(Bc / 16, Hc), T, 0, stream>>>(
      pooled, 2 * Dc, Hc * 2 * Dc, pExt, 32768, ext_b, HDc,
      zbuf, HDc, Hc * HDc);

  // 6. layernorm + relu
  ln_kernel<<<(int)((long)Bc * Hc / T), T, 0, stream>>>(zbuf, ln_g, ln_b, hemb);

  // 7. ctx = relu(mean_h(hemb) @ hub_W + hub_b)
  ctxmean_kernel<<<(int)((long)Bc * HDc / T), T, 0, stream>>>(hemb, ctxin);
  wmma_gemm_packed<HDc, 1><<<dim3(Bc / 16, 1), T, 0, stream>>>(
      ctxin, 0, HDc, pHub, 0, hub_b, 0, ctx, 0, HDc);

  // 8. pc = concat(hemb, ctx); hh = relu(pc @ pol_W1 + pol_b1) per head
  pc_kernel<<<(int)((long)Bc * Hc * 2 * HDc / T), T, 0, stream>>>(hemb, ctx, pc);
  wmma_gemm_packed<2 * HDc, 1><<<dim3(Bc / 16, Hc), T, 0, stream>>>(
      pc, 2 * HDc, Hc * 2 * HDc, pPol, 32768, pol_b1, HDc,
      hh, HDc, Hc * HDc);

  // 9. logits -> out[0 : B*H*A]
  logits_kernel<<<(int)(((long)Bc * Hc * Ac + T - 1) / T), T, 0, stream>>>(
      hh, pol_W2, pol_b2, out);

  // 10. value head -> out[B*H*A : B*H*A + B]
  gm_kernel<<<(int)((long)Bc * Dc / T), T, 0, stream>>>(he, gm);
  wmma_gemm_packed<Dc, 1><<<dim3(Bc / 16, 1), T, 0, stream>>>(
      gm, 0, Dc, pVal, 0, val_b1, 0, t1, 0, Dc);
  value_kernel<<<(Bc + T - 1) / T, T, 0, stream>>>(
      t1, val_W2, val_b2, out + (long)Bc * Hc * Ac);
}
